// MolecularGNN_85229331022353
// MI455X (gfx1250) — matrix operand order repaired and checked by reading, not compiled, not run on hardware
//
#include <hip/hip_runtime.h>
#include <hip/hip_bf16.h>

typedef __attribute__((ext_vector_type(16))) __bf16 v16bf;
typedef __attribute__((ext_vector_type(8)))  float  v8f;

#define DIM 512
#define NATOMS 32
#define BATCH 512
#define ROWS (BATCH * NATOMS)   // 16384
#define LHID 6
#define LOUT 6
#define HPAD 513                // LDS row stride (floats) for H tile
#define APAD 33                 // LDS row stride for adjacency

__device__ __forceinline__ unsigned short f2bf(float f) {
    unsigned u; __builtin_memcpy(&u, &f, 4);
    unsigned r = u + 0x7FFFu + ((u >> 16) & 1u);   // round-to-nearest-even
    return (unsigned short)(r >> 16);
}
__device__ __forceinline__ float bf2f(unsigned short h) {
    unsigned u = ((unsigned)h) << 16;
    float f; __builtin_memcpy(&f, &u, 4);
    return f;
}

union FragU { uint4 q[2]; v16bf v; };

// One K-step of operands: A fragment (16x32 bf16) + 8 B fragments (32x16 bf16)
struct KStep { FragU a; FragU b[8]; };

// ---------------------------------------------------------------------------
// Convert fp32 weights [L][K=512][N=512] to bf16 WMMA B-fragment layout:
// frag (kt,nt) covers K=kt*32..+31, N=nt*16..+15; per lane 16 bf16 contiguous.
// B lane layout: N = lane&15, K = (lane>>4)*16 + j   (j = 0..15)
// ---------------------------------------------------------------------------
__global__ void swizzle_w_kernel(const float* __restrict__ W,
                                 unsigned short* __restrict__ Wsw, int nLayers) {
    int idx = blockIdx.x * blockDim.x + threadIdx.x;
    if (idx >= nLayers * 16 * 32 * 32) return;
    int lane = idx & 31;
    int nt   = (idx >> 5) & 31;
    int kt   = (idx >> 10) & 15;
    int l    = idx >> 14;
    int n    = nt * 16 + (lane & 15);
    int kOff = (lane >> 4) * 16;
    const float* src = W + (size_t)l * DIM * DIM;
    unsigned short* dst = Wsw + (size_t)l * DIM * DIM + ((kt * 32 + nt) * 512 + lane * 16);
#pragma unroll
    for (int j = 0; j < 16; ++j) {
        int K = kt * 32 + kOff + j;
        dst[j] = f2bf(src[K * DIM + n]);
    }
}

// ---------------------------------------------------------------------------
// Embedding gather: V[row][d] = bf16(embed[fp[row]][d]),  row in [0,16384)
// ---------------------------------------------------------------------------
__global__ void gather_kernel(const int* __restrict__ fp,
                              const float* __restrict__ embed,
                              unsigned short* __restrict__ V) {
    int t = blockIdx.x * blockDim.x + threadIdx.x;   // one per 4 elements
    int base = t * 4;
    if (base >= ROWS * DIM) return;
    int row = base >> 9;
    int col = base & (DIM - 1);
    const float* src = embed + (size_t)fp[row] * DIM + col;
#pragma unroll
    for (int j = 0; j < 4; ++j) V[base + j] = f2bf(src[j]);
}

// ---------------------------------------------------------------------------
// Load all operands for K-step kt into registers (2 + 16 b128 loads).
// A frag lane layout (16-bit A 16x32): row = lane&15,
//   elems 0..15 -> K = kBase..kBase+15,
//   kBase = kt*32 + (lane>>4)*16.
// ---------------------------------------------------------------------------
__device__ __forceinline__ void load_kstep(KStep& s,
                                           const unsigned short* __restrict__ Arow,
                                           const unsigned short* __restrict__ Wfr,
                                           int kt) {
    int kBase = kt * 32 + ((threadIdx.x & 31) >> 4) * 16;
    s.a.q[0] = *(const uint4*)(Arow + kBase);
    s.a.q[1] = *(const uint4*)(Arow + kBase + 8);
    const unsigned short* wp = Wfr + (size_t)kt * 32 * 512;   // frag row for this kt
#pragma unroll
    for (int t = 0; t < 8; ++t) {
        const uint4* bp = (const uint4*)(wp + (size_t)t * 512);
        s.b[t].q[0] = bp[0];
        s.b[t].q[1] = bp[1];
    }
}

__device__ __forceinline__ void wmma_kstep(const KStep& s, v8f acc[8]) {
#pragma unroll
    for (int t = 0; t < 8; ++t)
        acc[t] = __builtin_amdgcn_wmma_f32_16x16x32_bf16(
            false, s.a.v, false, s.b[t].v, (short)0, acc[t], false, false);
}

// ---------------------------------------------------------------------------
// WMMA mainloop shared by both GEMMs, double-buffered over K-steps.
// Block = 256 threads = 8 waves. Output tile: 32 rows x 512 cols.
// Wave w: row-tile r = w>>2, col-tiles nt = (w&3)*8 .. +7.
// Accumulators start at zero; bias is added in the epilogue.
// ---------------------------------------------------------------------------
__device__ __forceinline__ void wmma_mainloop(const unsigned short* __restrict__ A,
                                              const unsigned short* __restrict__ Wsw,
                                              int rowBlock, v8f acc[8]) {
    const int lane  = threadIdx.x & 31;
    const int wave  = threadIdx.x >> 5;
    const int r     = wave >> 2;
    const int g     = wave & 3;
    const int laneRow = rowBlock * 32 + r * 16 + (lane & 15);
    const unsigned short* Arow = A + (size_t)laneRow * DIM;
    // per-lane base into swizzled weights for this wave's 8 col-tiles
    const unsigned short* Wfr = Wsw + ((size_t)(g * 8) * 512 + lane * 16);

    const v8f vzero = {0.f, 0.f, 0.f, 0.f, 0.f, 0.f, 0.f, 0.f};
#pragma unroll
    for (int t = 0; t < 8; ++t) acc[t] = vzero;

    KStep s0, s1;
    load_kstep(s0, Arow, Wfr, 0);
#pragma unroll
    for (int kt = 0; kt < 16; kt += 2) {
        load_kstep(s1, Arow, Wfr, kt + 1);
        wmma_kstep(s0, acc);
        if (kt + 2 < 16) load_kstep(s0, Arow, Wfr, kt + 2);
        wmma_kstep(s1, acc);
    }
}

// ---------------------------------------------------------------------------
// Feature layer (one molecule per block):
//   H = relu(V @ Wf + bf);  H2 = H + A*H;  Vout = H2 / max(||H2||_2, eps)
// Dynamic LDS: H[32][HPAD] + adj[32][APAD] + partial[8*32]
// ---------------------------------------------------------------------------
__global__ void layer_mp_kernel(const unsigned short* __restrict__ Vin,
                                const unsigned short* __restrict__ Wsw,
                                const float* __restrict__ bias,
                                const float* __restrict__ adj,
                                unsigned short* __restrict__ Vout) {
    extern __shared__ float smem[];
    float* Hlds    = smem;                         // 32*HPAD
    float* adjLds  = Hlds + 32 * HPAD;             // 32*APAD
    float* partial = adjLds + 32 * APAD;           // 256

    const int tid  = threadIdx.x;
    const int lane = tid & 31;
    const int wave = tid >> 5;
    const int mol  = blockIdx.x;

    v8f acc[8];
    wmma_mainloop(Vin, Wsw, mol, acc);

    // epilogue: bias + relu -> LDS
    {
        const int r = wave >> 2, g = wave & 3, kHalf = lane >> 4;
#pragma unroll
        for (int t = 0; t < 8; ++t) {
            int col = (g * 8 + t) * 16 + (lane & 15);
            float bv = bias[col];
#pragma unroll
            for (int e = 0; e < 8; ++e) {
                int row = r * 16 + kHalf * 8 + e;
                Hlds[row * HPAD + col] = fmaxf(acc[t][e] + bv, 0.0f);
            }
        }
    }

    // preload adjacency tile (32x32) into padded LDS
    const float* adjB = adj + (size_t)mol * NATOMS * NATOMS;
    for (int j = tid; j < NATOMS * NATOMS; j += 256)
        adjLds[(j >> 5) * APAD + (j & 31)] = adjB[j];
    __syncthreads();

    // message passing: thread (n = tid&31, dg = wave) handles d = dg*64..+63
    const int n  = tid & 31;
    const int dg = tid >> 5;
    float h2[64];
    float ss = 0.0f;
#pragma unroll 4
    for (int i = 0; i < 64; ++i) {
        int d = dg * 64 + i;
        float v = Hlds[n * HPAD + d];
#pragma unroll
        for (int m = 0; m < NATOMS; ++m)
            v = fmaf(adjLds[n * APAD + m], Hlds[m * HPAD + d], v);
        h2[i] = v;
        ss = fmaf(v, v, ss);
    }
    partial[dg * 32 + n] = ss;
    __syncthreads();

    float s = 0.0f;
#pragma unroll
    for (int p = 0; p < 8; ++p) s += partial[p * 32 + n];
    float inv = 1.0f / fmaxf(sqrtf(s), 1e-12f);

    // store normalized values back to LDS (MP reads all complete)
#pragma unroll 4
    for (int i = 0; i < 64; ++i)
        Hlds[n * HPAD + dg * 64 + i] = h2[i] * inv;
    __syncthreads();

    // coalesced bf16 writeout
    unsigned short* out = Vout + (size_t)mol * NATOMS * DIM;
#pragma unroll 4
    for (int i = 0; i < 64; ++i) {
        int f = i * 256 + tid;
        out[f] = f2bf(Hlds[(f >> 9) * HPAD + (f & (DIM - 1))]);
    }
}

// ---------------------------------------------------------------------------
// Plain GEMM + bias + relu (output MLP): out = relu(Min @ Wo + bo), bf16 out
// ---------------------------------------------------------------------------
__global__ void gemm_relu_kernel(const unsigned short* __restrict__ Min,
                                 const unsigned short* __restrict__ Wsw,
                                 const float* __restrict__ bias,
                                 unsigned short* __restrict__ Mout) {
    const int lane = threadIdx.x & 31;
    const int wave = threadIdx.x >> 5;
    v8f acc[8];
    wmma_mainloop(Min, Wsw, blockIdx.x, acc);
    const int r = wave >> 2, g = wave & 3, kHalf = lane >> 4;
#pragma unroll
    for (int t = 0; t < 8; ++t) {
        int col = (g * 8 + t) * 16 + (lane & 15);
        float bv = bias[col];
#pragma unroll
        for (int e = 0; e < 8; ++e) {
            int row = blockIdx.x * 32 + r * 16 + kHalf * 8 + e;
            Mout[(size_t)row * DIM + col] = f2bf(fmaxf(acc[t][e] + bv, 0.0f));
        }
    }
}

// ---------------------------------------------------------------------------
// Mean pool over atoms: Mol[b][d] = mean_n V[b*32+n][d]
// ---------------------------------------------------------------------------
__global__ void pool_kernel(const unsigned short* __restrict__ V,
                            unsigned short* __restrict__ Mol) {
    int idx = blockIdx.x * blockDim.x + threadIdx.x;
    if (idx >= BATCH * DIM) return;
    int b = idx >> 9;
    int d = idx & (DIM - 1);
    float s = 0.0f;
#pragma unroll 8
    for (int nn = 0; nn < NATOMS; ++nn)
        s += bf2f(V[((size_t)b * NATOMS + nn) * DIM + d]);
    Mol[idx] = f2bf(s * (1.0f / NATOMS));
}

// ---------------------------------------------------------------------------
// Final: out[b] = Mol[b] . Wp + bp   (one wave per row)
// ---------------------------------------------------------------------------
__global__ void final_dot_kernel(const unsigned short* __restrict__ Mol,
                                 const float* __restrict__ Wp,
                                 const float* __restrict__ bp,
                                 float* __restrict__ out) {
    const int lane = threadIdx.x & 31;
    const int wave = threadIdx.x >> 5;
    const int row  = blockIdx.x * 8 + wave;
    if (row >= BATCH) return;
    float s = 0.0f;
#pragma unroll
    for (int i = 0; i < 16; ++i) {
        int d = i * 32 + lane;
        s = fmaf(bf2f(Mol[(size_t)row * DIM + d]), Wp[d], s);
    }
#pragma unroll
    for (int off = 16; off >= 1; off >>= 1)
        s += __shfl_xor(s, off, 32);
    if (lane == 0) out[row] = s + bp[0];
}

// ---------------------------------------------------------------------------
extern "C" void kernel_launch(void* const* d_in, const int* in_sizes, int n_in,
                              void* d_out, int out_size, void* d_ws, size_t ws_size,
                              hipStream_t stream) {
    (void)in_sizes; (void)n_in; (void)out_size; (void)ws_size;
    const int*   fp    = (const int*)  d_in[0];
    const float* adj   = (const float*)d_in[1];
    const float* embed = (const float*)d_in[2];
    const float* Wf    = (const float*)d_in[3];
    const float* bf    = (const float*)d_in[4];
    const float* Wo    = (const float*)d_in[5];
    const float* bo    = (const float*)d_in[6];
    const float* Wp    = (const float*)d_in[7];
    const float* bp    = (const float*)d_in[8];
    float* out = (float*)d_out;

    char* ws = (char*)d_ws;
    size_t off = 0;
    auto carve = [&](size_t bytes) { char* p = ws + off; off += (bytes + 255) & ~(size_t)255; return p; };
    unsigned short* VactA = (unsigned short*)carve((size_t)ROWS * DIM * 2);
    unsigned short* VactB = (unsigned short*)carve((size_t)ROWS * DIM * 2);
    unsigned short* WfSw  = (unsigned short*)carve((size_t)LHID * DIM * DIM * 2);
    unsigned short* WoSw  = (unsigned short*)carve((size_t)LOUT * DIM * DIM * 2);
    unsigned short* MolA  = (unsigned short*)carve((size_t)BATCH * DIM * 2);
    unsigned short* MolB  = (unsigned short*)carve((size_t)BATCH * DIM * 2);

    // 1. weight swizzle fp32 -> bf16 WMMA-fragment layout
    swizzle_w_kernel<<<(LHID * 16 * 32 * 32 + 255) / 256, 256, 0, stream>>>(Wf, WfSw, LHID);
    swizzle_w_kernel<<<(LOUT * 16 * 32 * 32 + 255) / 256, 256, 0, stream>>>(Wo, WoSw, LOUT);

    // 2. embedding gather -> bf16 activations
    gather_kernel<<<(ROWS * DIM / 4 + 255) / 256, 256, 0, stream>>>(fp, embed, VactA);

    // 3. six fused feature layers (GEMM + message passing + L2 normalize)
    size_t smemBytes = (size_t)(32 * HPAD + 32 * APAD + 256) * 4;
    unsigned short* vin = VactA;
    unsigned short* vout = VactB;
    for (int l = 0; l < LHID; ++l) {
        layer_mp_kernel<<<BATCH, 256, smemBytes, stream>>>(
            vin, WfSw + (size_t)l * DIM * DIM, bf + (size_t)l * DIM, adj, vout);
        unsigned short* t = vin; vin = vout; vout = t;
    }

    // 4. mean pooling over atoms
    pool_kernel<<<(BATCH * DIM + 255) / 256, 256, 0, stream>>>(vin, MolA);

    // 5. output MLP
    unsigned short* min_ = MolA;
    unsigned short* mout = MolB;
    for (int l = 0; l < LOUT; ++l) {
        gemm_relu_kernel<<<BATCH / 32, 256, 0, stream>>>(
            min_, WoSw + (size_t)l * DIM * DIM, bo + (size_t)l * DIM, mout);
        unsigned short* t = min_; min_ = mout; mout = t;
    }

    // 6. final projection
    final_dot_kernel<<<BATCH / 8, 256, 0, stream>>>(min_, Wp, bp, out);
}
